// ImprovedFusionModel_88725434401068
// MI455X (gfx1250) — matrix operand order repaired and checked by reading, not compiled, not run on hardware
//
#include <hip/hip_runtime.h>
#include <hip/hip_bf16.h>
#include <math.h>

typedef __attribute__((ext_vector_type(16))) __bf16 v16bf;
typedef __attribute__((ext_vector_type(8)))  float  v8f;
typedef __attribute__((ext_vector_type(4)))  unsigned int u32x4;
typedef __attribute__((ext_vector_type(8)))  int i32x8;
typedef __attribute__((ext_vector_type(4)))  int i32x4;

// ---------------- problem dims ----------------
#define BB   64
#define TT   130
#define PPL  577
#define DINS 768
#define DD   768
#define HH   8
#define HD_  96
#define MT   128                 // text rows per batch (T-2)
#define MI   576                 // image rows per batch (P-1)
#define NTR  (BB*MT)             // 8192
#define NIR  (BB*MI)             // 36864
#define CK   3104                // combined width padded 3073 -> 3104 (mult 32)

// =====================================================================
//  TDM: DMA one 128-row x 32-col bf16 tile (row stride ld elements) from
//  global into LDS (rows packed 64B apart). D# bitfields per CDNA5 ISA
//  §8.3 (group0) / §8.4 (group1); 2D tensor -> groups 2/3 zero.
//  Issued by a single thread; tracked by TENSORcnt.
//  (6-arg builtin form: g0, g1, g2, g3, extra, cpol — clang-23 lane.)
// =====================================================================
__device__ __forceinline__ void tdm_load_tile_128x32(
    unsigned lds_byte_addr, const __bf16* gsrc, int ld)
{
    const unsigned long ga = (unsigned long)gsrc;
    u32x4 g0;
    g0[0] = 1u;                                        // count=1, user D#
    g0[1] = lds_byte_addr;                             // lds_addr [63:32]
    g0[2] = (unsigned)(ga & 0xffffffffu);              // global_addr lo
    g0[3] = (unsigned)((ga >> 32) & 0x1ffffffu)        // global_addr hi
          | (2u << 30);                                // type=2 ("image")
    const unsigned long s0 = (unsigned long)ld;        // dim0 stride (elems)
    i32x8 g1;
    g1[0] = (int)(1u << 16);                           // data_size=1 -> 2B
    g1[1] = (int)((32u & 0xffffu) << 16);              // tensor_dim0=32 lo16
    g1[2] = (int)(((32u >> 16) & 0xffffu)              // tensor_dim0 hi16
          | ((128u & 0xffffu) << 16));                 // tensor_dim1=128 lo16
    g1[3] = (int)(((128u >> 16) & 0xffffu)             // tensor_dim1 hi16
          | (32u << 16));                              // tile_dim0=32
    g1[4] = (int)(128u);                               // tile_dim1=128, td2=0
    g1[5] = (int)(s0 & 0xffffffffu);                   // dim0_stride lo32
    g1[6] = (int)((s0 >> 32) & 0xffffu);               // dim0_stride hi16
    g1[7] = 0;
    const i32x4 gz4 = {0, 0, 0, 0};
    const i32x8 gz8 = {0, 0, 0, 0, 0, 0, 0, 0};
    __builtin_amdgcn_tensor_load_to_lds(g0, g1, gz4, gz4, gz8, 0);
}

// =====================================================================
//  WMMA GEMM: C = A(bf16, MxK row-major) * B + bias
//    BT=true : B is NxK row-major (computes A*B^T)   -> attention scores
//    BT=false: B is KxN row-major                     -> weights / PV
//  Batched via blockIdx.z: zb = z/HJ, zh = z%HJ, element-offset strides.
//  Tile 128x128x32, 256 threads = 8 waves (2x4), each wave 64x32 via
//  4x2 v_wmma_f32_16x16x32_bf16 with f32 accumulation.
//  Staging: TDM tensor_load_to_lds for full aligned tiles (A always;
//  B when BT); 128-bit vector fast path otherwise; branchless clamped
//  scalar path at edges. NN B-transpose stays manual (TDM can't transpose).
// =====================================================================
template<bool BT, bool CF32>
__global__ __launch_bounds__(256) void gemm_bf16_wmma(
    const __bf16* __restrict__ A, int lda, long sAb, long sAh,
    const __bf16* __restrict__ Bm, int ldb, long sBb, long sBh,
    void* __restrict__ Cp, int ldc, long sCb, long sCh,
    const float* __restrict__ bias,
    int M, int N, int K, int HJ)
{
    __shared__ __bf16 As[128][32];   // [m][k]
    __shared__ __bf16 Bs[128][32];   // [n][k]

    const int zb = blockIdx.z / HJ, zh = blockIdx.z % HJ;
    const __bf16* Ab = A  + (long)zb*sAb + (long)zh*sAh;
    const __bf16* Bb = Bm + (long)zb*sBb + (long)zh*sBh;
    const long cOff  = (long)zb*sCb + (long)zh*sCh;

    const bool alnA = ((lda & 7) == 0) && ((((unsigned long)Ab) & 15) == 0);
    const bool alnB = ((ldb & 7) == 0) && ((((unsigned long)Bb) & 15) == 0);

    const int mb = blockIdx.y * 128, nb = blockIdx.x * 128;
    const int tid  = threadIdx.x;
    const int lane = tid & 31, wave = tid >> 5;
    const int wm = (wave >> 2) * 64;       // wave row base   (2 waves in M)
    const int wn = (wave & 3) * 32;        // wave col base   (4 waves in N)
    const int lr = lane & 15;              // lane row/col within 16
    const int lh = (lane >> 4) * 8;        // A K-octet select / C row-half
    const int kB = (lane >> 4) * 16;       // K-half offset; unused (B uses lh)

    // TDM usable when the whole 128x32 tile is in-bounds and aligned
    const bool tdmA = alnA && (mb + 128 <= M) && ((K & 31) == 0);
    const bool tdmB = BT && alnB && (nb + 128 <= N) && ((K & 31) == 0);

    v8f acc[4][2];
#pragma unroll
    for (int r = 0; r < 4; ++r)
#pragma unroll
        for (int c = 0; c < 2; ++c)
            acc[r][c] = (v8f){0.f,0.f,0.f,0.f,0.f,0.f,0.f,0.f};

    for (int kb = 0; kb < K; kb += 32) {
        // ---- TDM tile DMA (one issue per workgroup, thread 0) ----
        if (tdmA || tdmB) {
            if (tid == 0) {
                if (tdmA)
                    tdm_load_tile_128x32(
                        (unsigned)(unsigned long)&As[0][0],
                        Ab + (long)mb * lda + kb, lda);
                if (tdmB)
                    tdm_load_tile_128x32(
                        (unsigned)(unsigned long)&Bs[0][0],
                        Bb + (long)nb * ldb + kb, ldb);
                __builtin_amdgcn_s_wait_tensorcnt(0);
            }
        }
        // ---- stage A tile (128x32) when TDM not usable ----
        if (!tdmA) {
            const int m = tid >> 1, ks = (tid & 1) * 16;
            const int gm = mb + m;
            const __bf16* ap = Ab + (long)gm * lda + kb + ks;
            __bf16* dst = &As[m][ks];
            if (gm < M && (kb + ks + 16) <= K && alnA) {
                ((uint4*)dst)[0] = ((const uint4*)ap)[0];
                ((uint4*)dst)[1] = ((const uint4*)ap)[1];
            } else {
#pragma unroll
                for (int i = 0; i < 16; ++i) {
                    const int gk = kb + ks + i;
                    const bool ok = (gm < M) && (gk < K);
                    const __bf16 v = Ab[ok ? ((long)gm * lda + gk) : 0];
                    dst[i] = ok ? v : (__bf16)0.f;
                }
            }
        }
        // ---- stage B tile into [n][k] layout ----
        if (BT) {      // B row-major NxK: straight copy
            if (!tdmB) {
                const int n = tid >> 1, ks = (tid & 1) * 16;
                const int gn = nb + n;
                const __bf16* bp = Bb + (long)gn * ldb + kb + ks;
                __bf16* dst = &Bs[n][ks];
                if (gn < N && (kb + ks + 16) <= K && alnB) {
                    ((uint4*)dst)[0] = ((const uint4*)bp)[0];
                    ((uint4*)dst)[1] = ((const uint4*)bp)[1];
                } else {
#pragma unroll
                    for (int i = 0; i < 16; ++i) {
                        const int gk = kb + ks + i;
                        const bool ok = (gn < N) && (gk < K);
                        const __bf16 v = Bb[ok ? ((long)gn * ldb + gk) : 0];
                        dst[i] = ok ? v : (__bf16)0.f;
                    }
                }
            }
        } else {       // B row-major KxN: transpose while staging
            const int k = tid >> 3, ns = (tid & 7) * 16;
            const int gk = kb + k;
            const __bf16* bp = Bb + (long)gk * ldb + nb + ns;
            if (gk < K && (nb + ns + 16) <= N && alnB) {
                union { uint4 q[2]; __bf16 h[16]; } u;
                u.q[0] = ((const uint4*)bp)[0];
                u.q[1] = ((const uint4*)bp)[1];
#pragma unroll
                for (int i = 0; i < 16; ++i) Bs[ns + i][k] = u.h[i];
            } else {
#pragma unroll
                for (int i = 0; i < 16; ++i) {
                    const int gn = nb + ns + i;
                    const bool ok = (gk < K) && (gn < N);
                    const __bf16 v = Bb[ok ? ((long)gk * ldb + gn) : 0];
                    Bs[ns + i][k] = ok ? v : (__bf16)0.f;
                }
            }
        }
        __syncthreads();

        // ---- load fragments per ISA layout ----
        v16bf af[4], bfr[2];
#pragma unroll
        for (int r = 0; r < 4; ++r) {
            const int row = wm + r * 16 + lr;
#pragma unroll
            for (int i = 0; i < 8; ++i) {
                af[r][i]     = As[row][lh + i];          // K = lh..lh+7
                af[r][8 + i] = As[row][16 + lh + i];     // K = 16+lh..
            }
        }
#pragma unroll
        for (int c = 0; c < 2; ++c) {
            const int col = wn + c * 16 + lr;
#pragma unroll
            for (int i = 0; i < 8; ++i) {
                bfr[c][i]     = Bs[col][lh + i];         // K = lh..lh+7
                bfr[c][8 + i] = Bs[col][16 + lh + i];    // K = 16+lh..
            }
        }

#pragma unroll
        for (int r = 0; r < 4; ++r)
#pragma unroll
            for (int c = 0; c < 2; ++c)
                acc[r][c] = __builtin_amdgcn_wmma_f32_16x16x32_bf16(
                    false, af[r], false, bfr[c], (short)0, acc[r][c],
                    false, false);
        __syncthreads();
    }

    // ---- epilogue: C layout = VGPR i -> M = i + 8*(lane>=16), N = lr ----
#pragma unroll
    for (int r = 0; r < 4; ++r) {
#pragma unroll
        for (int c = 0; c < 2; ++c) {
            const int col = nb + wn + c * 16 + lr;
            if (col >= N) continue;
            const float badd = bias ? bias[col] : 0.f;
#pragma unroll
            for (int i = 0; i < 8; ++i) {
                const int row = mb + wm + r * 16 + lh + i;
                if (row >= M) continue;
                const float v = acc[r][c][i] + badd;
                const long off = cOff + (long)row * ldc + col;
                if constexpr (CF32) ((float*)Cp)[off] = v;
                else               ((__bf16*)Cp)[off] = (__bf16)v;
            }
        }
    }
}

// =====================================================================
//  elementwise / reduction kernels
// =====================================================================
__global__ void k_f32_to_bf16(const float* __restrict__ in,
                              __bf16* __restrict__ out, long n)
{
    long i = (long)blockIdx.x * blockDim.x + threadIdx.x;
    if (i < n) out[i] = (__bf16)in[i];
}

__global__ void k_zero_bf16(__bf16* __restrict__ out, long n)
{
    long i = (long)blockIdx.x * blockDim.x + threadIdx.x;
    if (i < n) out[i] = (__bf16)0.f;
}

// out[(b*rows+r)*DIN + d] = in[(b*srcSeq + rowOff + r)*DIN + d]  (f32->bf16)
__global__ void k_slice_bf16(const float* __restrict__ in,
                             __bf16* __restrict__ out,
                             int rows, int srcSeq, int rowOff, long n)
{
    long i = (long)blockIdx.x * blockDim.x + threadIdx.x;
    if (i >= n) return;
    int  d  = (int)(i % DINS);
    long r  = i / DINS;
    int  b  = (int)(r / rows);
    int  rr = (int)(r % rows);
    out[i] = (__bf16)in[((long)b * srcSeq + rowOff + rr) * DINS + d];
}

__global__ void k_add_bf16(const __bf16* __restrict__ a,
                           const __bf16* __restrict__ b,
                           __bf16* __restrict__ c, long n)
{
    long i = (long)blockIdx.x * blockDim.x + threadIdx.x;
    if (i < n) c[i] = (__bf16)((float)a[i] + (float)b[i]);
}

// layernorm over last dim (N<=768), optional residual + relu, in-place safe
__global__ __launch_bounds__(256) void k_layernorm(
    const __bf16* __restrict__ x, const __bf16* __restrict__ res,
    const float* __restrict__ g, const float* __restrict__ beta,
    __bf16* __restrict__ out, int N, int relu)
{
    const long row = blockIdx.x;
    const __bf16* xr = x + row * N;
    const __bf16* rr = res ? res + row * N : nullptr;
    __bf16* orow = out + row * N;
    __shared__ float red[256];
    const int tid = threadIdx.x;

    float lv[3];
    int cnt = 0;
    float s = 0.f;
    for (int d = tid; d < N; d += 256) {
        float v = (float)xr[d] + (rr ? (float)rr[d] : 0.f);
        lv[cnt++] = v; s += v;
    }
    red[tid] = s; __syncthreads();
    for (int st = 128; st > 0; st >>= 1) {
        if (tid < st) red[tid] += red[tid + st];
        __syncthreads();
    }
    const float mean = red[0] / (float)N; __syncthreads();
    float s2 = 0.f;
    for (int j = 0; j < cnt; ++j) { float d = lv[j] - mean; s2 += d * d; }
    red[tid] = s2; __syncthreads();
    for (int st = 128; st > 0; st >>= 1) {
        if (tid < st) red[tid] += red[tid + st];
        __syncthreads();
    }
    const float rstd = rsqrtf(red[0] / (float)N + 1e-5f);
    for (int d = tid, j = 0; d < N; d += 256, ++j) {
        float v = (lv[j] - mean) * rstd * g[d] + beta[d];
        if (relu) v = fmaxf(v, 0.f);
        orow[d] = (__bf16)v;
    }
}

// masked row softmax: scores f32 [z][M][N] -> probs bf16, scale folded in
__global__ __launch_bounds__(256) void k_softmax(
    const float* __restrict__ S, __bf16* __restrict__ P,
    const unsigned char* __restrict__ maskBase, int maskStride,
    int Mrows, int N, int H, float scale)
{
    const int m = blockIdx.x, z = blockIdx.y;
    const int zb = z / H;
    const float* row = S + ((long)z * Mrows + m) * N;
    __bf16*     prow = P + ((long)z * Mrows + m) * N;
    const unsigned char* mrow =
        maskBase ? maskBase + (long)zb * maskStride : nullptr;
    __shared__ float red[256];
    const int tid = threadIdx.x;

    float lv[3]; bool lok[3];
    int cnt = 0;
    float lmax = -3.0e38f;
    for (int j = tid; j < N; j += 256) {
        const bool ok = (!mrow) || (mrow[j] != 0);
        const float v = ok ? row[j] * scale : -3.0e38f;
        lv[cnt] = v; lok[cnt] = ok; ++cnt;
        lmax = fmaxf(lmax, v);
    }
    red[tid] = lmax; __syncthreads();
    for (int st = 128; st > 0; st >>= 1) {
        if (tid < st) red[tid] = fmaxf(red[tid], red[tid + st]);
        __syncthreads();
    }
    const float rmax = red[0]; __syncthreads();
    float ls = 0.f;
    for (int j = 0; j < cnt; ++j) {
        const float e = lok[j] ? __expf(lv[j] - rmax) : 0.f;
        lv[j] = e; ls += e;
    }
    red[tid] = ls; __syncthreads();
    for (int st = 128; st > 0; st >>= 1) {
        if (tid < st) red[tid] += red[tid + st];
        __syncthreads();
    }
    const float rinv = 1.f / red[0];
    for (int d = tid, j = 0; d < N; d += 256, ++j)
        prow[d] = (__bf16)(lv[j] * rinv);
}

// pooling: masked text avg, text max, img avg/max, conflict -> combined[CK]
__global__ __launch_bounds__(256) void k_pool(
    const __bf16* __restrict__ tf, const __bf16* __restrict__ imf,
    const unsigned char* __restrict__ am, __bf16* __restrict__ comb)
{
    const int b = blockIdx.x, tid = threadIdx.x;
    __shared__ float rd[256], rna[256], rnb[256];

    float cntf = 0.f;
    for (int j = 0; j < MT; ++j)
        if (am[(long)b * TT + 1 + j]) cntf += 1.f;
    if (cntf < 1.f) cntf = 1.f;

    float dot = 0.f, na2 = 0.f, nb2 = 0.f;
    for (int d = tid; d < DD; d += 256) {
        float ts = 0.f, tm = -3.0e38f;
        for (int j = 0; j < MT; ++j) {
            const float v = (float)tf[((long)b * MT + j) * DD + d];
            if (am[(long)b * TT + 1 + j]) ts += v;
            tm = fmaxf(tm, v);                 // text max is UNmasked
        }
        float is = 0.f, im = -3.0e38f;
        for (int j = 0; j < MI; ++j) {
            const float v = (float)imf[((long)b * MI + j) * DD + d];
            is += v; im = fmaxf(im, v);
        }
        const float ta = ts / cntf, ia = is / (float)MI;
        const long cb = (long)b * CK;
        comb[cb + d]        = (__bf16)ta;
        comb[cb + 768 + d]  = (__bf16)ia;
        comb[cb + 1536 + d] = (__bf16)tm;
        comb[cb + 2304 + d] = (__bf16)im;
        dot += ta * ia; na2 += ta * ta; nb2 += ia * ia;
    }
    rd[tid] = dot; rna[tid] = na2; rnb[tid] = nb2; __syncthreads();
    for (int st = 128; st > 0; st >>= 1) {
        if (tid < st) {
            rd[tid] += rd[tid + st];
            rna[tid] += rna[tid + st];
            rnb[tid] += rnb[tid + st];
        }
        __syncthreads();
    }
    if (tid == 0) {
        const float na = fmaxf(sqrtf(rna[0]), 1e-8f);
        const float nb = fmaxf(sqrtf(rnb[0]), 1e-8f);
        comb[(long)b * CK + 3072] = (__bf16)(1.f - rd[0] / (na * nb));
    }
}

// =====================================================================
//  host helpers
// =====================================================================
static void gemm_weight(hipStream_t stream, const __bf16* A, int M, int K,
                        int N, const __bf16* W, const float* bias, __bf16* C)
{
    dim3 g((N + 127) / 128, (M + 127) / 128, 1);
    gemm_bf16_wmma<false, false><<<g, 256, 0, stream>>>(
        A, K, 0, 0, W, N, 0, 0, (void*)C, N, 0, 0, bias, M, N, K, 1);
}

// attention: Q[b][Mq][D] (head cols h*96), K/V[b][Nk][D] -> ctx[b][Mq][D]
static void run_attention(hipStream_t stream,
                          const __bf16* Q, int Mq,
                          const __bf16* Kb, const __bf16* Vb, int Nk,
                          __bf16* ctx, float* scores, __bf16* probs,
                          const unsigned char* maskBase, int maskStride)
{
    const float scale = 1.0f / sqrtf((float)HD_);
    const int   BC = 2;                       // batch chunk (score buf bound)
    const long  sQb = (long)Mq * DD, sKb = (long)Nk * DD;
    for (int bb = 0; bb < BB; bb += BC) {
        const int Z = BC * HH;
        {   // scores = Q * K^T   (f32 out)
            dim3 g((Nk + 127) / 128, (Mq + 127) / 128, Z);
            gemm_bf16_wmma<true, true><<<g, 256, 0, stream>>>(
                Q + (long)bb * sQb, DD, sQb, HD_,
                Kb + (long)bb * sKb, DD, sKb, HD_,
                (void*)scores, Nk, (long)HH * Mq * Nk, (long)Mq * Nk,
                nullptr, Mq, Nk, HD_, HH);
        }
        {   // masked softmax -> bf16 probs
            const unsigned char* mb =
                maskBase ? maskBase + (long)bb * maskStride : nullptr;
            k_softmax<<<dim3(Mq, Z), 256, 0, stream>>>(
                scores, probs, mb, maskStride, Mq, Nk, HH, scale);
        }
        {   // ctx = P * V
            dim3 g(1, (Mq + 127) / 128, Z);
            gemm_bf16_wmma<false, false><<<g, 256, 0, stream>>>(
                probs, Nk, (long)HH * Mq * Nk, (long)Mq * Nk,
                Vb + (long)bb * sKb, DD, sKb, HD_,
                (void*)(ctx + (long)bb * sQb), DD, sQb, HD_,
                nullptr, Mq, HD_, Nk, HH);
        }
    }
}

// =====================================================================
extern "C" void kernel_launch(void* const* d_in, const int* in_sizes, int n_in,
                              void* d_out, int out_size, void* d_ws,
                              size_t ws_size, hipStream_t stream)
{
    (void)in_sizes; (void)n_in; (void)out_size; (void)ws_size;
    const float* text_hidden   = (const float*)d_in[0];
    const float* vision_hidden = (const float*)d_in[1];
    const unsigned char* amask = (const unsigned char*)d_in[2];
    auto Pf = [&](int i) { return (const float*)d_in[i]; };

    // ---- workspace carve (bump allocator, 256B aligned) ----
    char* base = (char*)d_ws; size_t off = 0;
    auto alloc = [&](size_t bytes) -> void* {
        void* p = base + off;
        off = (off + bytes + 255) & ~(size_t)255;
        return p;
    };
    auto abf = [&](long n) { return (__bf16*)alloc((size_t)n * 2); };

    const long WD = (long)DD * DD;
    __bf16 *wTproj = abf(WD), *wIproj = abf(WD), *wQt = abf(WD),
           *wKi = abf(WD), *wVi = abf(WD), *wQi = abf(WD), *wKt = abf(WD),
           *wVt = abf(WD), *wOt = abf(WD), *wOi = abf(WD),
           *wtq = abf(WD), *wtk = abf(WD), *wtv = abf(WD), *wto = abf(WD),
           *wiq = abf(WD), *wik = abf(WD), *wiv = abf(WD), *wio = abf(WD),
           *wc1 = abf((long)CK * 768), *wc2 = abf((long)768 * 256),
           *wc3 = abf(256);

    const long TN = (long)NTR * DD, INel = (long)NIR * DD;
    __bf16 *Tin = abf(TN), *T0 = abf(TN), *T1 = abf(TN), *T2 = abf(TN),
           *T3 = abf(TN), *T4 = abf(TN), *T5 = abf(TN);
    __bf16 *Iin = abf(INel), *I0 = abf(INel), *IA = abf(INel),
           *IB = abf(INel), *IC = abf(INel);
    float*  scores = (float*)alloc((size_t)2 * HH * MI * MI * 4);
    __bf16* probs  = abf((long)2 * HH * MI * MI);
    __bf16* comb   = abf((long)BB * CK);
    __bf16* h1     = abf((long)BB * DD);
    __bf16* h2     = abf((long)BB * 256);

    // ---- convert weights f32 -> bf16 ----
    struct WCV { int idx; long n; __bf16* dst; } wl[] = {
        {3,WD,wTproj},{7,WD,wIproj},{11,WD,wQt},{13,WD,wKi},{15,WD,wVi},
        {17,WD,wQi},{19,WD,wKt},{21,WD,wVt},{23,WD,wOt},{25,WD,wOi},
        {27,WD,wtq},{29,WD,wtk},{31,WD,wtv},{33,WD,wto},
        {35,WD,wiq},{37,WD,wik},{39,WD,wiv},{41,WD,wio},
        {47,(long)3073*768,wc1},{51,(long)768*256,wc2},{55,256,wc3}};
    for (auto& w : wl)
        k_f32_to_bf16<<<dim3((unsigned)((w.n + 255) / 256)), 256, 0, stream>>>(
            Pf(w.idx), w.dst, w.n);
    // zero-pad c1_w rows 3073..CK-1 so the classifier GEMM runs K=CK
    {
        const long padn = (long)(CK - 3073) * 768;
        k_zero_bf16<<<dim3((unsigned)((padn + 255) / 256)), 256, 0, stream>>>(
            wc1 + (long)3073 * 768, padn);
    }

    // ---- slice + convert inputs ----
    k_slice_bf16<<<dim3((unsigned)((TN + 255) / 256)), 256, 0, stream>>>(
        text_hidden, Tin, MT, TT, 1, TN);
    k_slice_bf16<<<dim3((unsigned)((INel + 255) / 256)), 256, 0, stream>>>(
        vision_hidden, Iin, MI, PPL, 1, INel);

    // ---- input projections + LN ----
    gemm_weight(stream, Tin, NTR, DINS, DD, wTproj, Pf(4), T5);
    k_layernorm<<<NTR, 256, 0, stream>>>(T5, nullptr, Pf(5), Pf(6), T0, DD, 0);
    gemm_weight(stream, Iin, NIR, DINS, DD, wIproj, Pf(8), IA);
    k_layernorm<<<NIR, 256, 0, stream>>>(IA, nullptr, Pf(9), Pf(10), I0, DD, 0);

    // ---- cross-attention Q/K/V (from pre-residual feats, per reference) ----
    gemm_weight(stream, T0, NTR, DD, DD, wQt, Pf(12), T1);   // q_text
    gemm_weight(stream, I0, NIR, DD, DD, wKi, Pf(14), IB);   // k_img
    gemm_weight(stream, I0, NIR, DD, DD, wVi, Pf(16), IC);   // v_img
    gemm_weight(stream, I0, NIR, DD, DD, wQi, Pf(18), IA);   // q_img
    gemm_weight(stream, T0, NTR, DD, DD, wKt, Pf(20), T2);   // k_text
    gemm_weight(stream, T0, NTR, DD, DD, wVt, Pf(22), T3);   // v_text

    // text cross (no mask) -> ctx T4 -> out proj T5
    run_attention(stream, T1, MT, IB, IC, MI, T4, scores, probs, nullptr, 0);
    gemm_weight(stream, T4, NTR, DD, DD, wOt, Pf(24), T5);   // text_cross

    // image cross (text key mask) -> ctx Iin (img_in dead) -> out proj IA
    run_attention(stream, IA, MI, T2, T3, MT, Iin, scores, probs,
                  amask + 1, TT);
    gemm_weight(stream, Iin, NIR, DD, DD, wOi, Pf(26), IA);  // img_cross

    // residuals
    k_add_bf16<<<dim3((unsigned)((TN + 255) / 256)), 256, 0, stream>>>(
        T0, T5, T0, TN);
    k_add_bf16<<<dim3((unsigned)((INel + 255) / 256)), 256, 0, stream>>>(
        I0, IA, I0, INel);

    // ---- text self-attention ----
    gemm_weight(stream, T0, NTR, DD, DD, wtq, Pf(28), T1);
    gemm_weight(stream, T0, NTR, DD, DD, wtk, Pf(30), T2);
    gemm_weight(stream, T0, NTR, DD, DD, wtv, Pf(32), T3);
    run_attention(stream, T1, MT, T2, T3, MT, T4, scores, probs,
                  amask + 1, TT);
    gemm_weight(stream, T4, NTR, DD, DD, wto, Pf(34), T5);
    k_layernorm<<<NTR, 256, 0, stream>>>(T0, T5, Pf(43), Pf(44), T0, DD, 0);

    // ---- image self-attention ----
    gemm_weight(stream, I0, NIR, DD, DD, wiq, Pf(36), IA);
    gemm_weight(stream, I0, NIR, DD, DD, wik, Pf(38), IB);
    gemm_weight(stream, I0, NIR, DD, DD, wiv, Pf(40), IC);
    run_attention(stream, IA, MI, IB, IC, MI, Iin, scores, probs, nullptr, 0);
    gemm_weight(stream, Iin, NIR, DD, DD, wio, Pf(42), IA);
    k_layernorm<<<NIR, 256, 0, stream>>>(I0, IA, Pf(45), Pf(46), I0, DD, 0);

    // ---- pooling -> combined [64][CK] (pre-zeroed for pad cols) ----
    {
        const long cn = (long)BB * CK;
        k_zero_bf16<<<dim3((unsigned)((cn + 255) / 256)), 256, 0, stream>>>(
            comb, cn);
    }
    k_pool<<<BB, 256, 0, stream>>>(T0, I0, amask, comb);

    // ---- classifier head (K padded to CK, fully vector path) ----
    gemm_weight(stream, comb, BB, CK, DD, wc1, Pf(48), h1);
    k_layernorm<<<BB, 256, 0, stream>>>(h1, nullptr, Pf(49), Pf(50), h1, DD, 1);
    gemm_weight(stream, h1, BB, DD, 256, wc2, Pf(52), h2);
    k_layernorm<<<BB, 256, 0, stream>>>(h2, nullptr, Pf(53), Pf(54), h2, 256, 1);

    // logits: [64,256] x [256,1] + c3_b -> d_out (f32)
    gemm_bf16_wmma<false, true><<<dim3(1, 1, 1), 256, 0, stream>>>(
        h2, 256, 0, 0, wc3, 1, 0, 0, d_out, 1, 0, 0, Pf(56), BB, 1, 256, 1);
}
